// ImprovedBothMamba_43954695307653
// MI455X (gfx1250) — matrix operand order repaired and checked by reading, not compiled, not run on hardware
//
#include <hip/hip_runtime.h>
#include <hip/hip_bf16.h>

// ---------------------------------------------------------------------------
// ImprovedBothMamba for MI455X (gfx1250, wave32, WMMA bf16 16x16x32)
// ---------------------------------------------------------------------------

typedef __attribute__((ext_vector_type(16))) __bf16 v16bf;
typedef __attribute__((ext_vector_type(8)))  float  v8f;

#define HW 12544           // 112*112
#define NPIX 25088         // B*H*W

// Fast transcendentals: v_rcp_f32 / v_exp_f32 / v_log_f32 (TRANS pipe,
// co-executes with WMMA) instead of IEEE div / libm log1p sequences.
__device__ __forceinline__ float fast_rcp(float x)  { return __builtin_amdgcn_rcpf(x); }
__device__ __forceinline__ float sigmoid_f(float x) { return fast_rcp(1.f + __expf(-x)); }
__device__ __forceinline__ float silu_f(float x)    { return x * sigmoid_f(x); }
__device__ __forceinline__ float softplus_f(float x){ return (x > 20.f) ? x : __logf(1.f + __expf(x)); }

__device__ __forceinline__ v8f wmma_bf16(v16bf a, v16bf b, v8f c) {
  // D = A(16x32 bf16) * B(32x16 bf16) + C(16x16 f32)
  return __builtin_amdgcn_wmma_f32_16x16x32_bf16(false, a, false, b, (short)0, c, false, false);
}

// A-fragment (16x32, bf16) from an LDS f32 tile [rows][ld]; zero-pad k >= kmax.
// Layout: lanes 0..15 / 16..31 both carry M=lane&15; K halves split per ISA table.
__device__ __forceinline__ v16bf load_a_lds(const float* lds, int ld, int m0, int k0,
                                            int kmax, int lane) {
  const int m  = m0 + (lane & 15);
  const int kb = (lane & 16) ? 8 : 0;
  v16bf a;
#pragma unroll
  for (int v = 0; v < 8; ++v) {
    const int kk = (v < 4) ? (kb + 2 * v) : (16 + kb + 2 * (v - 4));
#pragma unroll
    for (int q = 0; q < 2; ++q) {
      const int k = k0 + kk + q;
      a[2 * v + q] = (__bf16)((k < kmax) ? lds[m * ld + k] : 0.f);
    }
  }
  return a;
}

// B-fragment (32x16, bf16) from a global f32 weight, row-major [N][ldk].
// out[n] = sum_k act[k] * W[n][k]  -> B[k][n] = W[n][k].  Zero-pad OOB n/k.
// (Consecutive k reads get merged into global_load_b128 + v_cvt_pk_bf16_f32.)
__device__ __forceinline__ v16bf load_b_w(const float* __restrict__ W, int Nmax, int Kmax,
                                          int ldk, int n0, int k0, int lane) {
  const int n   = n0 + (lane & 15);
  const int kb  = (lane & 16) ? 8 : 0;
  const bool nok = (n < Nmax);
  v16bf b;
#pragma unroll
  for (int e = 0; e < 16; ++e) {
    const int k = k0 + ((e < 8) ? (kb + e) : (16 + kb + (e - 8)));
    b[e] = (__bf16)((nok && k < Kmax) ? W[n * ldk + k] : 0.f);
  }
  return b;
}

// ---------------------------------------------------------------------------
// Kernel 1: spatial mamba (L=1, d_model=96, d_inner=192, dt_rank=6)
//   L==1 collapses the scan: y_d = xc_d*(dt_d*<B,C> + D_d) * silu(z_d)
//   conv reduces to conv_b + conv_w[:,3]*xi.
// One wave per block; 2 tiles of 16 pixels per block (32 pixels, b constant).
// ---------------------------------------------------------------------------
__global__ __launch_bounds__(32)
void spa_mamba_kernel(const float* __restrict__ x,
                      const float* __restrict__ in_proj,   // [384][96]
                      const float* __restrict__ conv_w,    // [192][4]
                      const float* __restrict__ conv_b,    // [192]
                      const float* __restrict__ x_proj,    // [38][192]
                      const float* __restrict__ dt_w,      // [192][6]
                      const float* __restrict__ dt_b,      // [192]
                      const float* __restrict__ Dp,        // [192]
                      const float* __restrict__ out_proj,  // [96][192]
                      float* __restrict__ spa_y,           // [NPIX][96]
                      float* __restrict__ stats)           // raw sums
{
  __shared__ float s_xc[16 * 192];   // silu(conv(xi)); later overwritten with y
  __shared__ float s_sz[16 * 192];   // silu(z)
  __shared__ float s_proj[16 * 48];  // x_proj output (38 valid, padded cols are 0)
  __shared__ float s_bc[16];
  __shared__ float s_sum[4];
  __shared__ float s_ssq[4];

  const int lane = threadIdx.x;
  const int pix_base = blockIdx.x * 32;
  const int b = pix_base / HW;
  const int hw_wg = pix_base - b * HW;
  const float* xb = x + (size_t)b * 96 * HW;

  if (lane < 4) { s_sum[lane] = 0.f; s_ssq[lane] = 0.f; }

  for (int mt = 0; mt < 2; ++mt) {
    const int hw0 = hw_wg + mt * 16;
    __syncthreads();

    // ---- GEMM1: xz = x_tile(16x96) @ in_proj^T(96x384), fused conv+silu ----
    for (int nt = 0; nt < 24; ++nt) {
      v8f c = {};
#pragma unroll
      for (int kt = 0; kt < 3; ++kt) {
        // A-frag straight from global x (channel-major): loop-invariant over nt,
        // the compiler hoists all three K-tiles into VGPRs.
        const int m  = lane & 15;
        const int kb = (lane & 16) ? 8 : 0;
        v16bf a;
#pragma unroll
        for (int v = 0; v < 8; ++v) {
          const int kk = (v < 4) ? (kb + 2 * v) : (16 + kb + 2 * (v - 4));
          const int ch = kt * 32 + kk;
          a[2 * v]     = (__bf16)xb[(size_t)ch * HW + hw0 + m];
          a[2 * v + 1] = (__bf16)xb[(size_t)(ch + 1) * HW + hw0 + m];
        }
        const v16bf bf = load_b_w(in_proj, 384, 96, 96, nt * 16, kt * 32, lane);
        c = wmma_bf16(a, bf, c);
      }
      const int col = nt * 16 + (lane & 15);
      const int mh  = (lane >> 4) << 3;
#pragma unroll
      for (int r = 0; r < 8; ++r) {
        const int m = r + mh;
        const float v = c[r];
        if (col < 192) {  // xi -> depthwise causal conv at L=1 -> silu
          s_xc[m * 192 + col] = silu_f(conv_b[col] + conv_w[col * 4 + 3] * v);
        } else {          // z -> silu
          s_sz[m * 192 + (col - 192)] = silu_f(v);
        }
      }
    }
    __syncthreads();

    // ---- GEMM2: proj = xc(16x192) @ x_proj^T(192x38, N padded to 48) ----
    for (int nt = 0; nt < 3; ++nt) {
      v8f c = {};
#pragma unroll
      for (int kt = 0; kt < 6; ++kt) {
        const v16bf a  = load_a_lds(s_xc, 192, 0, kt * 32, 192, lane);
        const v16bf bf = load_b_w(x_proj, 38, 192, 192, nt * 16, kt * 32, lane);
        c = wmma_bf16(a, bf, c);
      }
      const int col = nt * 16 + (lane & 15);
      const int mh  = (lane >> 4) << 3;
#pragma unroll
      for (int r = 0; r < 8; ++r) s_proj[(r + mh) * 48 + col] = c[r];
    }
    __syncthreads();

    // ---- <B,C> dot per pixel ----
    if (lane < 16) {
      float acc = 0.f;
#pragma unroll
      for (int s = 0; s < 16; ++s)
        acc += s_proj[lane * 48 + 6 + s] * s_proj[lane * 48 + 22 + s];
      s_bc[lane] = acc;
    }
    __syncthreads();

    // ---- dt (rank 6) + collapsed scan + gate; y written in-place into s_xc ----
    for (int idx = lane; idx < 16 * 192; idx += 32) {
      const int m = idx / 192;
      const int d = idx - m * 192;
      float acc = dt_b[d];
#pragma unroll
      for (int r = 0; r < 6; ++r) acc += s_proj[m * 48 + r] * dt_w[d * 6 + r];
      const float dt = softplus_f(acc);
      const float xc = s_xc[idx];
      s_xc[idx] = xc * (dt * s_bc[m] + Dp[d]) * s_sz[idx];
    }
    __syncthreads();

    // ---- GEMM3: out = y(16x192) @ out_proj^T(192x96) + GN partial sums ----
    for (int nt = 0; nt < 6; ++nt) {
      v8f c = {};
#pragma unroll
      for (int kt = 0; kt < 6; ++kt) {
        const v16bf a  = load_a_lds(s_xc, 192, 0, kt * 32, 192, lane);
        const v16bf bf = load_b_w(out_proj, 96, 192, 192, nt * 16, kt * 32, lane);
        c = wmma_bf16(a, bf, c);
      }
      const int col = nt * 16 + (lane & 15);
      const int mh  = (lane >> 4) << 3;
      const int g   = col / 24;
      float lsum = 0.f, lssq = 0.f;
#pragma unroll
      for (int r = 0; r < 8; ++r) {
        const float v = c[r];
        spa_y[(size_t)(pix_base + mt * 16 + r + mh) * 96 + col] = v;
        lsum += v; lssq += v * v;
      }
      atomicAdd(&s_sum[g], lsum);
      atomicAdd(&s_ssq[g], lssq);
    }
  }
  __syncthreads();
  if (lane < 4) {
    atomicAdd(&stats[(b * 4 + lane) * 2 + 0], s_sum[lane]);
    atomicAdd(&stats[(b * 4 + lane) * 2 + 1], s_ssq[lane]);
  }
}

// ---------------------------------------------------------------------------
// Kernel 2: spectral mamba (L=4 tokens of 24 ch; d_inner=48, dt_rank=2)
// 16 pixels per block = 64 token rows; real 4-step scan with 16-wide state.
// ---------------------------------------------------------------------------
__global__ __launch_bounds__(32)
void spe_mamba_kernel(const float* __restrict__ x,
                      const float* __restrict__ in_proj,   // [96][24]
                      const float* __restrict__ conv_w,    // [48][4]
                      const float* __restrict__ conv_b,    // [48]
                      const float* __restrict__ x_proj,    // [34][48]
                      const float* __restrict__ dt_w,      // [48][2]
                      const float* __restrict__ dt_b,      // [48]
                      const float* __restrict__ A_log,     // [48][16]
                      const float* __restrict__ Dp,        // [48]
                      const float* __restrict__ out_proj,  // [24][48]
                      float* __restrict__ spe_y,           // [NPIX][96]
                      float* __restrict__ stats)
{
  __shared__ float s_xz[64 * 96];    // xz; cols 0..47 later reused for y
  __shared__ float s_xc[64 * 48];
  __shared__ float s_proj[64 * 48];  // 34 valid, padded cols are 0
  __shared__ float s_A[48 * 16];
  __shared__ float s_sum[4];
  __shared__ float s_ssq[4];

  const int lane = threadIdx.x;
  const int pix_base = blockIdx.x * 16;
  const int b = pix_base / HW;
  const int hw0 = pix_base - b * HW;
  const float* xb = x + (size_t)b * 96 * HW;

  if (lane < 4) { s_sum[lane] = 0.f; s_ssq[lane] = 0.f; }
  for (int i = lane; i < 48 * 16; i += 32) s_A[i] = -__expf(A_log[i]);
  __syncthreads();

  // ---- GEMM1: xz(64x96) = xrow(64x24, K pad 32) @ in_proj^T ----
  for (int mt = 0; mt < 4; ++mt) {
    const int r0 = mt * 16;
    const int m  = lane & 15;
    const int row = r0 + m;
    const int p = row >> 2, t = row & 3;
    const int kb = (lane & 16) ? 8 : 0;
    v16bf a;
#pragma unroll
    for (int v = 0; v < 8; ++v) {
      const int kk = (v < 4) ? (kb + 2 * v) : (16 + kb + 2 * (v - 4));
#pragma unroll
      for (int q = 0; q < 2; ++q) {
        const int k = kk + q;
        a[2 * v + q] = (__bf16)((k < 24) ? xb[(size_t)(t * 24 + k) * HW + hw0 + p] : 0.f);
      }
    }
    for (int nt = 0; nt < 6; ++nt) {
      const v16bf bf = load_b_w(in_proj, 96, 24, 24, nt * 16, 0, lane);
      v8f c = {};
      c = wmma_bf16(a, bf, c);
      const int col = nt * 16 + (lane & 15);
      const int mh  = (lane >> 4) << 3;
#pragma unroll
      for (int r = 0; r < 8; ++r) s_xz[(r0 + r + mh) * 96 + col] = c[r];
    }
  }
  __syncthreads();

  // ---- causal depthwise conv over 4 tokens + silu ----
  for (int idx = lane; idx < 64 * 48; idx += 32) {
    const int r = idx / 48, d = idx - r * 48;
    const int p = r >> 2, t = r & 3;
    float acc = conv_b[d];
#pragma unroll
    for (int k = 0; k < 4; ++k) {
      const int tt = t + k - 3;
      if (tt >= 0) acc += conv_w[d * 4 + k] * s_xz[(p * 4 + tt) * 96 + d];
    }
    s_xc[idx] = silu_f(acc);
  }
  __syncthreads();

  // ---- GEMM2: proj(64x34 padN48) = xc(64x48 padK64) @ x_proj^T ----
  for (int mt = 0; mt < 4; ++mt) {
    for (int nt = 0; nt < 3; ++nt) {
      v8f c = {};
#pragma unroll
      for (int kt = 0; kt < 2; ++kt) {
        const v16bf a  = load_a_lds(s_xc, 48, mt * 16, kt * 32, 48, lane);
        const v16bf bf = load_b_w(x_proj, 34, 48, 48, nt * 16, kt * 32, lane);
        c = wmma_bf16(a, bf, c);
      }
      const int col = nt * 16 + (lane & 15);
      const int mh  = (lane >> 4) << 3;
#pragma unroll
      for (int r = 0; r < 8; ++r) s_proj[(mt * 16 + r + mh) * 48 + col] = c[r];
    }
  }
  __syncthreads();

  // ---- selective scan: each lane walks (pixel,d) pairs; y -> s_xz cols 0..47 ----
  for (int idx = lane; idx < 16 * 48; idx += 32) {
    const int p = idx / 48, d = idx - p * 48;
    float Av[16];
#pragma unroll
    for (int s = 0; s < 16; ++s) Av[s] = s_A[d * 16 + s];
    const float w0 = dt_w[d * 2], w1 = dt_w[d * 2 + 1], bb = dt_b[d], Dd = Dp[d];
    float h[16];
#pragma unroll
    for (int s = 0; s < 16; ++s) h[s] = 0.f;
#pragma unroll
    for (int t = 0; t < 4; ++t) {
      const int r = p * 4 + t;
      const float dt = softplus_f(s_proj[r * 48] * w0 + s_proj[r * 48 + 1] * w1 + bb);
      const float xc = s_xc[r * 48 + d];
      const float dtxc = dt * xc;
      float y = 0.f;
#pragma unroll
      for (int s = 0; s < 16; ++s) {
        const float dA = __expf(dt * Av[s]);
        h[s] = dA * h[s] + dtxc * s_proj[r * 48 + 2 + s];   // B
        y += h[s] * s_proj[r * 48 + 18 + s];                // C
      }
      y += xc * Dd;
      y *= silu_f(s_xz[r * 96 + 48 + d]);                   // gate with silu(z)
      s_xz[r * 96 + d] = y;
    }
  }
  __syncthreads();

  // ---- GEMM3: out(64x24 padN32) = y(64x48 padK64) @ out_proj^T + stats ----
  float ls[4] = {0.f, 0.f, 0.f, 0.f}, lq[4] = {0.f, 0.f, 0.f, 0.f};
  for (int mt = 0; mt < 4; ++mt) {
    for (int nt = 0; nt < 2; ++nt) {
      v8f c = {};
#pragma unroll
      for (int kt = 0; kt < 2; ++kt) {
        const v16bf a  = load_a_lds(s_xz, 96, mt * 16, kt * 32, 48, lane);
        const v16bf bf = load_b_w(out_proj, 24, 48, 48, nt * 16, kt * 32, lane);
        c = wmma_bf16(a, bf, c);
      }
      const int col = nt * 16 + (lane & 15);
      if (col < 24) {
        const int mh = (lane >> 4) << 3;
#pragma unroll
        for (int r = 0; r < 8; ++r) {
          const int row = mt * 16 + r + mh;
          const int p = row >> 2, t = row & 3;   // channel group == token index
          const float v = c[r];
          spe_y[(size_t)(pix_base + p) * 96 + t * 24 + col] = v;
          ls[r & 3] += v; lq[r & 3] += v * v;    // (r&3)==t since mh%4==0
        }
      }
    }
  }
#pragma unroll
  for (int g = 0; g < 4; ++g) { atomicAdd(&s_sum[g], ls[g]); atomicAdd(&s_ssq[g], lq[g]); }
  __syncthreads();
  if (lane < 4) {
    atomicAdd(&stats[(8 + b * 4 + lane) * 2 + 0], s_sum[lane]);
    atomicAdd(&stats[(8 + b * 4 + lane) * 2 + 1], s_ssq[lane]);
  }
}

// ---------------------------------------------------------------------------
// Kernel 3: finalize GroupNorm stats (16 groups total: 8 spa + 8 spe)
// ---------------------------------------------------------------------------
__global__ void finalize_stats_kernel(const float* __restrict__ raw, float* __restrict__ fin) {
  const int i = threadIdx.x;
  if (i < 16) {
    const float cnt = 24.f * (float)HW;  // 24 channels * 12544 pixels per (b,g)
    const float mean = raw[i * 2] / cnt;
    const float var  = raw[i * 2 + 1] / cnt - mean * mean;
    fin[i * 2]     = mean;
    fin[i * 2 + 1] = rsqrtf(var + 1e-5f);
  }
}

// ---------------------------------------------------------------------------
// Kernel 4: GN + silu + residual for both branches, then fusion gate.
// ---------------------------------------------------------------------------
__global__ __launch_bounds__(32)
void fusion_kernel(const float* __restrict__ x,
                   const float* __restrict__ spa_y, const float* __restrict__ spe_y,
                   const float* __restrict__ spa_g, const float* __restrict__ spa_b,
                   const float* __restrict__ spe_g, const float* __restrict__ spe_b,
                   const float* __restrict__ fin,
                   const float* __restrict__ w1,  // [96][192]
                   const float* __restrict__ b1,  // [96]
                   const float* __restrict__ w2,  // [96]
                   const float* __restrict__ b2,  // [1]
                   float* __restrict__ out)
{
  __shared__ float s_f[16 * 192];   // [spa(96) | spe(96)] per pixel
  __shared__ float s_a[16 * 96];
  __shared__ float s_alpha[16];

  const int lane = threadIdx.x;
  const int pix_base = blockIdx.x * 32;
  const int b = pix_base / HW;
  const int hw_wg = pix_base - b * HW;
  const float* xb = x + (size_t)b * 96 * HW;
  float* ob = out + (size_t)b * 96 * HW;

  for (int mt = 0; mt < 2; ++mt) {
    const int hw0 = hw_wg + mt * 16;
    const int pi0 = pix_base + mt * 16;
    __syncthreads();

    // ---- GroupNorm + silu + residual -> concat tile ----
    for (int idx = lane; idx < 16 * 192; idx += 32) {
      const int m = idx / 192, col = idx - m * 192;
      const int isspe = (col >= 96);
      const int ch = isspe ? (col - 96) : col;
      const int g = ch / 24;
      const int si = ((isspe ? 8 : 0) + b * 4 + g) * 2;
      const float* yb = isspe ? spe_y : spa_y;
      float v = yb[(size_t)(pi0 + m) * 96 + ch];
      v = (v - fin[si]) * fin[si + 1] * (isspe ? spe_g[ch] : spa_g[ch])
          + (isspe ? spe_b[ch] : spa_b[ch]);
      s_f[idx] = silu_f(v) + xb[(size_t)ch * HW + hw0 + m];
    }
    __syncthreads();

    // ---- a = silu(f(16x192) @ w1^T(192x96) + b1) ----
    for (int nt = 0; nt < 6; ++nt) {
      v8f c = {};
#pragma unroll
      for (int kt = 0; kt < 6; ++kt) {
        const v16bf a  = load_a_lds(s_f, 192, 0, kt * 32, 192, lane);
        const v16bf bf = load_b_w(w1, 96, 192, 192, nt * 16, kt * 32, lane);
        c = wmma_bf16(a, bf, c);
      }
      const int col = nt * 16 + (lane & 15);
      const int mh  = (lane >> 4) << 3;
      const float bias = b1[col];
#pragma unroll
      for (int r = 0; r < 8; ++r) s_a[(r + mh) * 96 + col] = silu_f(c[r] + bias);
    }
    __syncthreads();

    // ---- alpha = sigmoid(a . w2 + b2) per pixel ----
    if (lane < 16) {
      float acc = b2[0];
      for (int cch = 0; cch < 96; ++cch) acc += s_a[lane * 96 + cch] * w2[cch];
      s_alpha[lane] = sigmoid_f(acc);
    }
    __syncthreads();

    // ---- blend + store (B,C,H,W) ----
    for (int idx = lane; idx < 16 * 96; idx += 32) {
      const int m = idx / 96, ch = idx - m * 96;
      const float al = s_alpha[m];
      ob[(size_t)ch * HW + hw0 + m] =
          s_f[m * 192 + ch] * al + s_f[m * 192 + 96 + ch] * (1.f - al);
    }
  }
}

// ---------------------------------------------------------------------------
// Host launcher
// ---------------------------------------------------------------------------
extern "C" void kernel_launch(void* const* d_in, const int* in_sizes, int n_in,
                              void* d_out, int out_size, void* d_ws, size_t ws_size,
                              hipStream_t stream) {
  (void)in_sizes; (void)n_in; (void)out_size; (void)ws_size;

  const float* x        = (const float*)d_in[0];
  // spa_params.mamba
  const float* spa_inp  = (const float*)d_in[1];   // in_proj [384][96]
  const float* spa_cw   = (const float*)d_in[2];   // conv_w  [192][4]
  const float* spa_cb   = (const float*)d_in[3];   // conv_b  [192]
  const float* spa_xp   = (const float*)d_in[4];   // x_proj  [38][192]
  const float* spa_dtw  = (const float*)d_in[5];   // dt_proj_w [192][6]
  const float* spa_dtb  = (const float*)d_in[6];   // dt_proj_b [192]
  // d_in[7] = spa A_log: unused (L==1 collapses the scan exactly)
  const float* spa_D    = (const float*)d_in[8];   // D [192]
  const float* spa_outp = (const float*)d_in[9];   // out_proj [96][192]
  const float* spa_g    = (const float*)d_in[10];  // gn gamma [96]
  const float* spa_bb   = (const float*)d_in[11];  // gn beta  [96]
  // spe_params.mamba
  const float* spe_inp  = (const float*)d_in[12];  // in_proj [96][24]
  const float* spe_cw   = (const float*)d_in[13];  // conv_w  [48][4]
  const float* spe_cb   = (const float*)d_in[14];  // conv_b  [48]
  const float* spe_xp   = (const float*)d_in[15];  // x_proj  [34][48]
  const float* spe_dtw  = (const float*)d_in[16];  // dt_proj_w [48][2]
  const float* spe_dtb  = (const float*)d_in[17];  // dt_proj_b [48]
  const float* spe_A    = (const float*)d_in[18];  // A_log [48][16]
  const float* spe_D    = (const float*)d_in[19];  // D [48]
  const float* spe_outp = (const float*)d_in[20];  // out_proj [24][48]
  const float* spe_g    = (const float*)d_in[21];  // gn gamma [96]
  const float* spe_bb   = (const float*)d_in[22];  // gn beta  [96]
  // fus_params
  const float* fw1      = (const float*)d_in[23];  // [96][192]
  const float* fb1      = (const float*)d_in[24];  // [96]
  const float* fw2      = (const float*)d_in[25];  // [96] (shape (1,96))
  const float* fb2      = (const float*)d_in[26];  // [1]

  float* ws    = (float*)d_ws;
  float* spa_y = ws;                         // NPIX*96 floats
  float* spe_y = ws + (size_t)NPIX * 96;     // NPIX*96 floats
  float* raw   = ws + (size_t)2 * NPIX * 96; // 32 floats (sum, sumsq) x16 groups
  float* fin   = raw + 32;                   // 32 floats (mean, rstd) x16 groups

  hipMemsetAsync(raw, 0, 32 * sizeof(float), stream);

  spa_mamba_kernel<<<NPIX / 32, 32, 0, stream>>>(
      x, spa_inp, spa_cw, spa_cb, spa_xp, spa_dtw, spa_dtb, spa_D, spa_outp,
      spa_y, raw);

  spe_mamba_kernel<<<NPIX / 16, 32, 0, stream>>>(
      x, spe_inp, spe_cw, spe_cb, spe_xp, spe_dtw, spe_dtb, spe_A, spe_D, spe_outp,
      spe_y, raw);

  finalize_stats_kernel<<<1, 32, 0, stream>>>(raw, fin);

  fusion_kernel<<<NPIX / 32, 32, 0, stream>>>(
      x, spa_y, spe_y, spa_g, spa_bb, spe_g, spe_bb, fin,
      fw1, fb1, fw2, fb2, (float*)d_out);
}
